// CurvatureAttention_50448685859295
// MI455X (gfx1250) — matrix operand order repaired and checked by reading, not compiled, not run on hardware
//
#include <hip/hip_runtime.h>

// ---- problem constants (from reference) ----
#define NN    4096   // nodes
#define INC   256    // input channels
#define OUTC  256    // output channels
#define NH    8      // heads
#define DD    32     // dims per head
#define CFD   6      // curvature feature dim

typedef __attribute__((ext_vector_type(16))) __bf16 v16bf;
typedef __attribute__((ext_vector_type(8)))  __bf16 v8bf;
typedef __attribute__((ext_vector_type(8)))  float  v8f;

// =====================================================================
// Kernel 1: h = x @ W^T  via v_wmma_f32_16x16x32_bf16, one 16x16 tile
// per wave. Result stored TRANSPOSED as bf16: h_t[c][n], so that the
// attention kernel's B-fragments are two contiguous 16B loads.
// A-frag (16x32 bf16), lane L: row M=L%16; k = half*8+e (e<8), 16+half*8+(e-8)
// B-frag (32x16 bf16), lane L: col N=L%16; k = half*8+e (e<8), 16+half*8+(e-8)
// C/D    (16x16 f32),  lane L: col N=L%16; row M = v + 8*half
// =====================================================================
__global__ __launch_bounds__(32)
void k_gemm_h(const float* __restrict__ x, const float* __restrict__ W,
              __bf16* __restrict__ h_t) {
  const int lane = threadIdx.x;
  const int half = lane >> 4;
  const int l16  = lane & 15;
  const int m = blockIdx.x * 16 + l16;   // node row (A)
  const int c = blockIdx.y * 16 + l16;   // out channel (B col)

  const float* xrow = x + (size_t)m * INC;
  const float* wrow = W + (size_t)c * INC;

  v8f acc = {};
#pragma unroll
  for (int kk = 0; kk < INC; kk += 32) {
    // ---- A fragment: x[m][k] ----
    float af[16];
    {
      float4 A0 = *(const float4*)(xrow + kk + half * 8);
      float4 A1 = *(const float4*)(xrow + kk + half * 8 + 4);
      float4 A2 = *(const float4*)(xrow + kk + 16 + half * 8);
      float4 A3 = *(const float4*)(xrow + kk + 16 + half * 8 + 4);
      af[0]=A0.x; af[1]=A0.y; af[2]=A0.z;  af[3]=A0.w;
      af[4]=A1.x; af[5]=A1.y; af[6]=A1.z;  af[7]=A1.w;
      af[8]=A2.x; af[9]=A2.y; af[10]=A2.z; af[11]=A2.w;
      af[12]=A3.x;af[13]=A3.y;af[14]=A3.z; af[15]=A3.w;
    }
    v16bf a;
#pragma unroll
    for (int e = 0; e < 16; ++e) a[e] = (__bf16)af[e];

    // ---- B fragment: W^T[k][c] = W[c][k], k = kk + half*8 + e (e<8), kk + 16 + half*8 + (e-8) ----
    float bf[16];
    {
      float4 B0 = *(const float4*)(wrow + kk + half * 8);
      float4 B1 = *(const float4*)(wrow + kk + half * 8 + 4);
      float4 B2 = *(const float4*)(wrow + kk + 16 + half * 8);
      float4 B3 = *(const float4*)(wrow + kk + 16 + half * 8 + 4);
      bf[0]=B0.x; bf[1]=B0.y; bf[2]=B0.z;  bf[3]=B0.w;
      bf[4]=B1.x; bf[5]=B1.y; bf[6]=B1.z;  bf[7]=B1.w;
      bf[8]=B2.x; bf[9]=B2.y; bf[10]=B2.z; bf[11]=B2.w;
      bf[12]=B3.x;bf[13]=B3.y;bf[14]=B3.z; bf[15]=B3.w;
    }
    v16bf b;
#pragma unroll
    for (int e = 0; e < 16; ++e) b[e] = (__bf16)bf[e];

    acc = __builtin_amdgcn_wmma_f32_16x16x32_bf16(false, a, false, b,
                                                  (short)0, acc, false, false);
  }

  // store transposed: h_t[c][node]
#pragma unroll
  for (int v = 0; v < 8; ++v) {
    int node = blockIdx.x * 16 + v + half * 8;
    h_t[(size_t)c * NN + node] = (__bf16)acc[v];
  }
}

// =====================================================================
// Kernel 2: e_l[n,h] = <h[n,h,:], a_l[h,:]>, e_r likewise.
// Reads h_t coalesced across n; stores transposed e_*_t[h][n].
// =====================================================================
__global__ __launch_bounds__(256)
void k_el_er(const __bf16* __restrict__ h_t,
             const float* __restrict__ a_l, const float* __restrict__ a_r,
             float* __restrict__ e_l_t, float* __restrict__ e_r_t) {
  const int n = blockIdx.x * 256 + threadIdx.x;
#pragma unroll
  for (int hh = 0; hh < NH; ++hh) {
    float sl = 0.f, sr = 0.f;
#pragma unroll
    for (int d = 0; d < DD; ++d) {
      const int c = hh * DD + d;
      const float v = (float)h_t[(size_t)c * NN + n];
      sl += v * a_l[c];
      sr += v * a_r[c];
    }
    e_l_t[hh * NN + n] = sl;
    e_r_t[hh * NN + n] = sr;
  }
}

// =====================================================================
// Kernel 3: fused masked-softmax attention (flash style).
// Grid: NN/16 workgroups x 256 threads (8 waves). Wave w = head w.
// Each wave owns out[i0:i0+16, head, 0:32]; streams j in tiles of 32.
// adj / curv tiles staged once per WG in LDS (shared by all 8 heads).
// Logits are built directly in WMMA A-fragment layout, so the online
// softmax row reduction is lane-local + one shfl_xor(16).
// =====================================================================
__global__ __launch_bounds__(256)
void k_attn(const float* __restrict__ e_l_t, const float* __restrict__ e_r_t,
            const __bf16* __restrict__ h_t, const int* __restrict__ adj,
            const float* __restrict__ cf, const float* __restrict__ Wc,
            const float* __restrict__ gatep, float* __restrict__ out) {
  __shared__ int   s_adj[16 * 32];
  __shared__ float s_cf[32 * CFD];

  const int tid  = threadIdx.x;
  const int hh   = tid >> 5;        // wave id == head
  const int lane = tid & 31;
  const int half = lane >> 4;
  const int l16  = lane & 15;
  const int i0   = blockIdx.x * 16;
  const int i    = i0 + l16;        // A-layout row for this lane

  const float gate = *gatep;
  float wc[CFD], cfi[CFD];
#pragma unroll
  for (int f = 0; f < CFD; ++f) wc[f]  = Wc[hh * CFD + f];
#pragma unroll
  for (int f = 0; f < CFD; ++f) cfi[f] = cf[(size_t)i * CFD + f];
  const float eli = e_l_t[hh * NN + i];

  v8f acc0 = {}, acc1 = {};
  float m_run = -1e30f, l_run = 0.f;

  for (int j0 = 0; j0 < NN; j0 += 32) {
    // ---- cooperative stage: adj 16x32 tile + cf 32x6 tile ----
    __syncthreads();
#pragma unroll
    for (int r = 0; r < 2; ++r) {
      const int idx = tid + r * 256;
      const int row = idx >> 5, col = idx & 31;
      s_adj[idx] = adj[(size_t)(i0 + row) * NN + j0 + col];
    }
    if (tid < 32 * CFD) {
      const int jj = tid / CFD, f = tid - jj * CFD;
      s_cf[tid] = cf[(size_t)(j0 + jj) * CFD + f];
    }
    if (j0 + 32 < NN)  // hint next adj tile toward the caches
      __builtin_prefetch(adj + (size_t)(i0 + (tid >> 5)) * NN + j0 + 32 + (tid & 31), 0, 1);
    __syncthreads();

    // ---- e_r for this lane's 16 k-slots (two runs of 8 contiguous j) ----
    float er16[16];
    {
      const float* base = e_r_t + hh * NN + j0;
      float4 E0 = *(const float4*)(base + half * 8);
      float4 E1 = *(const float4*)(base + half * 8 + 4);
      float4 E2 = *(const float4*)(base + 16 + half * 8);
      float4 E3 = *(const float4*)(base + 16 + half * 8 + 4);
      er16[0]=E0.x; er16[1]=E0.y; er16[2]=E0.z;  er16[3]=E0.w;
      er16[4]=E1.x; er16[5]=E1.y; er16[6]=E1.z;  er16[7]=E1.w;
      er16[8]=E2.x; er16[9]=E2.y; er16[10]=E2.z; er16[11]=E2.w;
      er16[12]=E3.x;er16[13]=E3.y;er16[14]=E3.z; er16[15]=E3.w;
    }

    // ---- logits in A-fragment slots ----
    float s[16];
#pragma unroll
    for (int e = 0; e < 16; ++e) {
      const int k = (e < 8) ? (half * 8 + e) : (16 + half * 8 + (e - 8));
      float v = eli + er16[e];
      v = (v > 0.f) ? v : 0.2f * v;                 // leaky_relu
      float bias = 0.f;
#pragma unroll
      for (int f = 0; f < CFD; ++f)
        bias += fabsf(cfi[f] - s_cf[k * CFD + f]) * wc[f];
      v += gate * bias;
      s[e] = (s_adj[l16 * 32 + k] > 0) ? v : -1e9f; // mask
    }

    // ---- online softmax update ----
    float mt = s[0];
#pragma unroll
    for (int e = 1; e < 16; ++e) mt = fmaxf(mt, s[e]);
    mt = fmaxf(mt, __shfl_xor(mt, 16));
    const float m_new = fmaxf(m_run, mt);
    const float scale = __expf(m_run - m_new);

    float psum = 0.f;
    v16bf pa;
#pragma unroll
    for (int e = 0; e < 16; ++e) {
      const float p = __expf(s[e] - m_new);
      psum += p;
      pa[e] = (__bf16)p;
    }
    psum += __shfl_xor(psum, 16);
    l_run = l_run * scale + psum;
    m_run = m_new;

    // rescale accumulators (row M = v + 8*half; state lives in lane M)
#pragma unroll
    for (int v = 0; v < 8; ++v) {
      const float sv = __shfl(scale, v + half * 8);
      acc0[v] *= sv;
      acc1[v] *= sv;
    }

    // ---- B fragments from transposed bf16 h: h_t[d][j], contiguous ----
    const int jb = j0 + half * 8;
    const __bf16* hr0 = h_t + (size_t)(hh * DD + 0  + l16) * NN + jb;
    const __bf16* hr1 = h_t + (size_t)(hh * DD + 16 + l16) * NN + jb;
    v8bf t0a = *(const v8bf*)hr0, t0b = *(const v8bf*)(hr0 + 16);
    v8bf t1a = *(const v8bf*)hr1, t1b = *(const v8bf*)(hr1 + 16);
    v16bf b0, b1;
#pragma unroll
    for (int e = 0; e < 8; ++e) {
      b0[e] = t0a[e]; b0[8 + e] = t0b[e];
      b1[e] = t1a[e]; b1[8 + e] = t1b[e];
    }

    acc0 = __builtin_amdgcn_wmma_f32_16x16x32_bf16(false, pa, false, b0,
                                                   (short)0, acc0, false, false);
    acc1 = __builtin_amdgcn_wmma_f32_16x16x32_bf16(false, pa, false, b1,
                                                   (short)0, acc1, false, false);
  }

  // ---- finalize: divide by row sum, write out[i][hh*32 + d] ----
#pragma unroll
  for (int v = 0; v < 8; ++v) {
    const float lv  = __shfl(l_run, v + half * 8);
    const float inv = 1.0f / lv;
    const int irow  = i0 + v + half * 8;
    out[(size_t)irow * OUTC + hh * DD + l16]      = acc0[v] * inv;
    out[(size_t)irow * OUTC + hh * DD + 16 + l16] = acc1[v] * inv;
  }
}

// =====================================================================
extern "C" void kernel_launch(void* const* d_in, const int* in_sizes, int n_in,
                              void* d_out, int out_size, void* d_ws, size_t ws_size,
                              hipStream_t stream) {
  const float* x    = (const float*)d_in[0];
  // d_in[1] = positions (unused by reference)
  const float* cf   = (const float*)d_in[2];
  const int*   adj  = (const int*)  d_in[3];
  const float* W    = (const float*)d_in[4];
  const float* a_l  = (const float*)d_in[5];
  const float* a_r  = (const float*)d_in[6];
  const float* Wc   = (const float*)d_in[7];
  const float* gate = (const float*)d_in[8];
  float* out = (float*)d_out;

  // workspace layout: h_t (bf16, OUTC x NN) | e_l_t | e_r_t (f32, NH x NN)
  __bf16* h_t  = (__bf16*)d_ws;
  float* e_l_t = (float*)((char*)d_ws + (size_t)OUTC * NN * sizeof(__bf16));
  float* e_r_t = e_l_t + (size_t)NH * NN;

  k_gemm_h<<<dim3(NN / 16, OUTC / 16), dim3(32), 0, stream>>>(x, W, h_t);
  k_el_er<<<dim3(NN / 256), dim3(256), 0, stream>>>(h_t, a_l, a_r, e_l_t, e_r_t);
  k_attn<<<dim3(NN / 16), dim3(256), 0, stream>>>(e_l_t, e_r_t, h_t, adj, cf,
                                                  Wc, gate, out);
}
